// BiasedAxialAttention_58548994179791
// MI455X (gfx1250) — matrix operand order repaired and checked by reading, not compiled, not run on hardware
//
#include <hip/hip_runtime.h>
#include <hip/hip_bf16.h>

// ---------------------------------------------------------------------------
// Biased axial attention for MI455X (gfx1250), wave32 + v_wmma_f32_16x16x32_f16.
// All GEMMs on the matrix pipes in f16 with f32 accumulation; LN/softmax/
// sigmoid in f32 (memory-bound stages).  ~55 GFLOP vs ~450 MB traffic:
// balanced against 23.3 TB/s HBM only with f16 WMMA on the matmuls.
// QK^T contraction (K=12288) is split into 4 chunks for occupancy; partials
// are reduced (with the learned bias term) inside the softmax kernel, keeping
// results bit-deterministic (no float atomics).
// ---------------------------------------------------------------------------

#define LSEQ 384
#define DP   128
#define NH   4
#define DH   32
#define NK   (LSEQ * DH)      // 12288 contraction for QK^T
#define NROW (LSEQ * LSEQ)    // 147456
#define KCH  4                // K-chunks for QK^T
#define KCL  (NK / KCH)       // 3072 per chunk

typedef __attribute__((ext_vector_type(16))) _Float16 v16h;
typedef __attribute__((ext_vector_type(8)))  _Float16 v8h;
typedef __attribute__((ext_vector_type(8)))  float    v8f;

// ---- WMMA fragment helpers (layouts per CDNA5 ISA 7.12.2, wave32) ----------

// A: 16x32 f16, M x K.  lane: m = m0 + (lane&15), group g = lane>>4.
// halves 0..7  <- k = k0 + g*8 + [0..8)
// halves 8..15 <- k = k0 + 16 + g*8 + [0..8)
__device__ __forceinline__ v16h load_a_frag(const _Float16* __restrict__ A,
                                            int lda, int m0, int k0, int lane) {
  int m = m0 + (lane & 15);
  int g = lane >> 4;
  const _Float16* p = A + (size_t)m * lda + k0 + g * 8;
  union { v16h v; v8h h[2]; } u;
  u.h[0] = *(const v8h*)(p);
  u.h[1] = *(const v8h*)(p + 16);
  return u.v;
}

// B supplied as B^T rows ([N][K] storage): lane n = n0 + (lane&15), g = lane>>4,
// halves 0..7 <- k = k0 + g*8 + [0..8), halves 8..15 <- k = k0 + 16 + g*8 + [0..8)
__device__ __forceinline__ v16h load_b_frag(const _Float16* __restrict__ BT,
                                            int ldb, int n0, int k0, int lane) {
  int nn = n0 + (lane & 15);
  int g  = lane >> 4;
  const _Float16* p = BT + (size_t)nn * ldb + k0 + g * 8;
  union { v16h v; v8h h[2]; } u;
  u.h[0] = *(const v8h*)(p);
  u.h[1] = *(const v8h*)(p + 16);
  return u.v;
}

__device__ __forceinline__ v8f wmma16(v16h a, v16h b, v8f c) {
  return __builtin_amdgcn_wmma_f32_16x16x32_f16(false, a, false, b,
                                                (short)0, c, false, false);
}

#define V8F_ZERO {0.f,0.f,0.f,0.f,0.f,0.f,0.f,0.f}

// ---- P0: transpose all five 128x128 weights to f16 [out][in] ---------------
__global__ void prep_weights_kernel(const float* __restrict__ Wq,
                                    const float* __restrict__ Wk,
                                    const float* __restrict__ Wv,
                                    const float* __restrict__ Wg,
                                    const float* __restrict__ Wout,
                                    _Float16* __restrict__ wt) {
  int t = blockIdx.x * blockDim.x + threadIdx.x;   // 16384
  if (t >= DP * DP) return;
  int o = t >> 7, ii = t & 127;
  int src = ii * DP + o;
  wt[0 * 16384 + t] = (_Float16)Wq[src];
  wt[1 * 16384 + t] = (_Float16)Wk[src];
  wt[2 * 16384 + t] = (_Float16)Wv[src];
  wt[3 * 16384 + t] = (_Float16)Wg[src];
  wt[4 * 16384 + t] = (_Float16)Wout[src];
}

// ---- P1: LN over pair (transposed frame): row r=(n,i) reads pair[i,n,:] ----
__global__ __launch_bounds__(128)
void ln_pair_kernel(const float* __restrict__ pair,
                    const float* __restrict__ gamma,
                    const float* __restrict__ beta,
                    _Float16* __restrict__ pn16) {
  int r = blockIdx.x;               // r = n*LSEQ + i
  int n = r / LSEQ, i = r - n * LSEQ;
  int c = threadIdx.x;
  float x = pair[((size_t)i * LSEQ + n) * DP + c];
  __shared__ float s1[128], s2[128];
  s1[c] = x; s2[c] = x * x;
  __syncthreads();
  for (int off = 64; off; off >>= 1) {
    if (c < off) { s1[c] += s1[c + off]; s2[c] += s2[c + off]; }
    __syncthreads();
  }
  float m = s1[0] * (1.0f / 128.0f);
  float v = s2[0] * (1.0f / 128.0f) - m * m;
  float y = (x - m) * rsqrtf(v + 1e-5f) * gamma[c] + beta[c];
  pn16[(size_t)r * DP + c] = (_Float16)y;
}

// ---- P1b: LN(bias) @ Wb -> bterm[h][i][j] ----------------------------------
// bias row (a,b): bias[0,a,b,:] == bias_t[0,b,a,:]  =>  i=b, j=a
__global__ __launch_bounds__(128)
void ln_bias_kernel(const float* __restrict__ bias,
                    const float* __restrict__ gamma,
                    const float* __restrict__ beta,
                    const float* __restrict__ Wb,
                    float* __restrict__ bterm) {
  int r = blockIdx.x;
  int a = r / LSEQ, b = r - a * LSEQ;
  int c = threadIdx.x;
  float x = bias[(size_t)r * DP + c];
  __shared__ float s1[128], s2[128];
  s1[c] = x; s2[c] = x * x;
  __syncthreads();
  for (int off = 64; off; off >>= 1) {
    if (c < off) { s1[c] += s1[c + off]; s2[c] += s2[c + off]; }
    __syncthreads();
  }
  float m = s1[0] * (1.0f / 128.0f);
  float v = s2[0] * (1.0f / 128.0f) - m * m;
  float y = (x - m) * rsqrtf(v + 1e-5f) * gamma[c] + beta[c];
  __shared__ float lns[128];
  lns[c] = y;
  __syncthreads();
  if (c < NH) {
    float s = 0.f;
    for (int t = 0; t < DP; ++t) s += lns[t] * Wb[t * NH + c];
    bterm[((size_t)c * LSEQ + b) * LSEQ + a] = s;   // [h][i][j]
  }
}

// ---- P2: projections [147456x128]@[128x512]; wave w owns Wq/Wk/Wv/Wg -------
__global__ __launch_bounds__(128)
void proj_kernel(const _Float16* __restrict__ pn16,
                 const _Float16* __restrict__ wt,
                 const float* __restrict__ bg,
                 _Float16* __restrict__ qT,     // [h][i][n*32+k]
                 _Float16* __restrict__ kT,     // [h][j][n*32+k]
                 _Float16* __restrict__ vT,     // [n][h][d][j]
                 _Float16* __restrict__ gate16) // [row][c]
{
  int wave = threadIdx.x >> 5;
  int lane = threadIdx.x & 31;
  int wv   = __builtin_amdgcn_readfirstlane(wave);   // scalar branch selector
  int r0 = blockIdx.x * 16;
  const _Float16* W = wt + (size_t)wave * 16384;

  v16h af[4];
#pragma unroll
  for (int kb = 0; kb < 4; ++kb) af[kb] = load_a_frag(pn16, DP, r0, kb * 32, lane);

  int g = lane >> 4, nn = lane & 15;
#pragma unroll
  for (int nt = 0; nt < 8; ++nt) {
    v8f acc = V8F_ZERO;
#pragma unroll
    for (int kb = 0; kb < 4; ++kb) {
      v16h bf = load_b_frag(W, DP, nt * 16, kb * 32, lane);
      acc = wmma16(af[kb], bf, acc);
    }
    int cc = nt * 16 + nn;
#pragma unroll
    for (int e = 0; e < 8; ++e) {
      int row = r0 + e + g * 8;
      int n = row / LSEQ, i = row - n * LSEQ;
      float val = acc[e];
      if (wv == 0) {              // Q, scaled 1/sqrt(32)
        int h = cc >> 5, k = cc & 31;
        qT[((size_t)(h * LSEQ + i)) * NK + n * 32 + k] =
            (_Float16)(val * 0.17677669529663687f);
      } else if (wv == 1) {       // K, scaled 1/L
        int h = cc >> 5, k = cc & 31;
        kT[((size_t)(h * LSEQ + i)) * NK + n * 32 + k] =
            (_Float16)(val * (1.0f / 384.0f));
      } else if (wv == 2) {       // V, transposed to [n][h][d][j], j = i
        int h = cc >> 5, d = cc & 31;
        vT[((size_t)((n * NH + h) * DH + d)) * LSEQ + i] = (_Float16)val;
      } else {                    // gate = sigmoid(x + bg), fast rcp
        float e1 = __expf(-(val + bg[cc]));
        float s  = __builtin_amdgcn_rcpf(1.0f + e1);
        gate16[(size_t)row * DP + cc] = (_Float16)s;
      }
    }
  }
}

// ---- S2: partial attn: attnp[ch][h][i][j] = Q_h @ K_h^T over K-chunk ch ----
// 64x64 block tile, 32x32 per wave, 2x2 accumulators; K chunk = 3072.
__global__ __launch_bounds__(128)
void attn_kernel(const _Float16* __restrict__ qT,
                 const _Float16* __restrict__ kT,
                 float* __restrict__ attnp) {
  int z = blockIdx.z;              // z = h*KCH + ch
  int h = z >> 2, ch = z & (KCH - 1);
  int wave = threadIdx.x >> 5, lane = threadIdx.x & 31;
  int i_base = blockIdx.x * 64 + (wave >> 1) * 32;
  int j_base = blockIdx.y * 64 + (wave & 1) * 32;
  const _Float16* Q = qT + (size_t)h * LSEQ * NK;
  const _Float16* K = kT + (size_t)h * LSEQ * NK;
  int k_begin = ch * KCL, k_end = k_begin + KCL;

  v8f acc00 = V8F_ZERO, acc01 = V8F_ZERO, acc10 = V8F_ZERO, acc11 = V8F_ZERO;
  for (int kk = k_begin; kk < k_end; kk += 32) {
    if (kk + 256 < k_end) { // uniform speculative prefetch -> global_prefetch_b8
      __builtin_prefetch(Q + (size_t)(i_base + (lane & 15)) * NK + kk + 256, 0, 1);
      __builtin_prefetch(K + (size_t)(j_base + (lane & 15)) * NK + kk + 256, 0, 1);
    }
    v16h a0 = load_a_frag(Q, NK, i_base,      kk, lane);
    v16h a1 = load_a_frag(Q, NK, i_base + 16, kk, lane);
    v16h b0 = load_b_frag(K, NK, j_base,      kk, lane);
    v16h b1 = load_b_frag(K, NK, j_base + 16, kk, lane);
    acc00 = wmma16(a0, b0, acc00);
    acc01 = wmma16(a0, b1, acc01);
    acc10 = wmma16(a1, b0, acc10);
    acc11 = wmma16(a1, b1, acc11);
  }
  int g = lane >> 4, nn = lane & 15;
  float* P = attnp + (size_t)(ch * NH + h) * LSEQ * LSEQ;
#pragma unroll
  for (int e = 0; e < 8; ++e) {
    int m = e + g * 8;
    {
      size_t idx = (size_t)(i_base + m) * LSEQ + j_base + nn;
      P[idx]      = acc00[e];
      P[idx + 16] = acc01[e];
    }
    {
      size_t idx = (size_t)(i_base + 16 + m) * LSEQ + j_base + nn;
      P[idx]      = acc10[e];
      P[idx + 16] = acc11[e];
    }
  }
}

// ---- S3: reduce 4 partials + bterm, softmax over j per (h,i) ---------------
__global__ __launch_bounds__(128)
void softmax_kernel(const float* __restrict__ attnp,
                    const float* __restrict__ bterm,
                    _Float16* __restrict__ a16) {
  int r = blockIdx.x;                       // h*LSEQ + i
  int t = threadIdx.x;
  const size_t rb = (size_t)r * LSEQ;
  const size_t cstride = (size_t)NH * LSEQ * LSEQ;   // chunk stride
  float x0 = bterm[rb + t], x1 = bterm[rb + t + 128], x2 = bterm[rb + t + 256];
#pragma unroll
  for (int ch = 0; ch < KCH; ++ch) {
    const float* P = attnp + ch * cstride + rb;
    x0 += P[t]; x1 += P[t + 128]; x2 += P[t + 256];
  }
  __shared__ float s[128];
  s[t] = fmaxf(x0, fmaxf(x1, x2));
  __syncthreads();
  for (int off = 64; off; off >>= 1) {
    if (t < off) s[t] = fmaxf(s[t], s[t + off]);
    __syncthreads();
  }
  float mx = s[0];
  __syncthreads();
  float e0 = __expf(x0 - mx), e1 = __expf(x1 - mx), e2 = __expf(x2 - mx);
  s[t] = e0 + e1 + e2;
  __syncthreads();
  for (int off = 64; off; off >>= 1) {
    if (t < off) s[t] += s[t + off];
    __syncthreads();
  }
  float inv = __builtin_amdgcn_rcpf(s[0]);   // v_rcp_f32, ~1 ulp
  a16[rb + t]       = (_Float16)(e0 * inv);
  a16[rb + t + 128] = (_Float16)(e1 * inv);
  a16[rb + t + 256] = (_Float16)(e2 * inv);
}

// ---- S4: O = A_h @ V_{n,h} [384x384]@[384x32], gated, -> og16 --------------
__global__ __launch_bounds__(128)
void av_kernel(const _Float16* __restrict__ a16,
               const _Float16* __restrict__ vT,
               const _Float16* __restrict__ gate16,
               _Float16* __restrict__ og16) {
  int n = blockIdx.x, h = blockIdx.y;
  int wave = threadIdx.x >> 5, lane = threadIdx.x & 31;
  const _Float16* A = a16 + (size_t)h * LSEQ * LSEQ;
  const _Float16* V = vT + (size_t)(n * NH + h) * DH * LSEQ;   // [d][j]

  v8f acc[6][2];
#pragma unroll
  for (int t = 0; t < 6; ++t) { acc[t][0] = (v8f)V8F_ZERO; acc[t][1] = (v8f)V8F_ZERO; }

  for (int kb = 0; kb < 12; ++kb) {
    v16h b0 = load_b_frag(V, LSEQ, 0,  kb * 32, lane);
    v16h b1 = load_b_frag(V, LSEQ, 16, kb * 32, lane);
#pragma unroll
    for (int t = 0; t < 6; ++t) {
      v16h a = load_a_frag(A, LSEQ, (wave * 6 + t) * 16, kb * 32, lane);
      acc[t][0] = wmma16(a, b0, acc[t][0]);
      acc[t][1] = wmma16(a, b1, acc[t][1]);
    }
  }
  int g = lane >> 4, nn = lane & 15;
#pragma unroll
  for (int t = 0; t < 6; ++t) {
#pragma unroll
    for (int e = 0; e < 8; ++e) {
      int i = (wave * 6 + t) * 16 + e + g * 8;
      size_t base = ((size_t)(n * LSEQ + i)) * DP + h * DH;
      float g0 = (float)gate16[base + nn];
      float g1 = (float)gate16[base + 16 + nn];
      og16[base + nn]      = (_Float16)(acc[t][0][e] * g0);
      og16[base + 16 + nn] = (_Float16)(acc[t][1][e] * g1);
    }
  }
}

// ---- S5: out = og @ Wout + bout, undo transpose ----------------------------
__global__ __launch_bounds__(128)
void out_kernel(const _Float16* __restrict__ og16,
                const _Float16* __restrict__ WoutT,
                const float* __restrict__ bout,
                float* __restrict__ out) {
  int wave = threadIdx.x >> 5, lane = threadIdx.x & 31;
  int r0 = (blockIdx.x * 4 + wave) * 16;
  v16h af[4];
#pragma unroll
  for (int kb = 0; kb < 4; ++kb) af[kb] = load_a_frag(og16, DP, r0, kb * 32, lane);
  int g = lane >> 4, nn = lane & 15;
#pragma unroll
  for (int nt = 0; nt < 8; ++nt) {
    v8f acc = V8F_ZERO;
#pragma unroll
    for (int kb = 0; kb < 4; ++kb) {
      v16h bf = load_b_frag(WoutT, DP, nt * 16, kb * 32, lane);
      acc = wmma16(af[kb], bf, acc);
    }
    int cc = nt * 16 + nn;
    float bb = bout[cc];
#pragma unroll
    for (int e = 0; e < 8; ++e) {
      int row = r0 + e + g * 8;          // row = n*LSEQ + i
      int n = row / LSEQ, i = row - n * LSEQ;
      out[((size_t)i * LSEQ + n) * DP + cc] = acc[e] + bb;   // out[i][n][c]
    }
  }
}

// ---------------------------------------------------------------------------
extern "C" void kernel_launch(void* const* d_in, const int* in_sizes, int n_in,
                              void* d_out, int out_size, void* d_ws, size_t ws_size,
                              hipStream_t stream) {
  (void)in_sizes; (void)n_in; (void)out_size; (void)ws_size;
  const float* pair    = (const float*)d_in[0];
  const float* bias    = (const float*)d_in[1];
  const float* gamma_p = (const float*)d_in[2];
  const float* beta_p  = (const float*)d_in[3];
  const float* gamma_b = (const float*)d_in[4];
  const float* beta_b  = (const float*)d_in[5];
  const float* Wq   = (const float*)d_in[6];
  const float* Wk   = (const float*)d_in[7];
  const float* Wv   = (const float*)d_in[8];
  const float* Wb   = (const float*)d_in[9];
  const float* Wg   = (const float*)d_in[10];
  const float* bg   = (const float*)d_in[11];
  const float* Wout = (const float*)d_in[12];
  const float* bout = (const float*)d_in[13];
  float* out = (float*)d_out;

  char* p = (char*)d_ws;
  auto take = [&](size_t bytes) {
    void* q = (void*)p;
    p += (bytes + 255) & ~(size_t)255;
    return q;
  };
  const size_t HBYTES = (size_t)NROW * DP * sizeof(_Float16);   // 37.75 MB
  _Float16* pn16   = (_Float16*)take(HBYTES);   // reused as og16 after proj
  _Float16* qT     = (_Float16*)take(HBYTES);
  _Float16* kT     = (_Float16*)take(HBYTES);
  _Float16* vT     = (_Float16*)take(HBYTES);
  _Float16* gate16 = (_Float16*)take(HBYTES);
  float*    bterm  = (float*)take((size_t)NH * LSEQ * LSEQ * sizeof(float));
  float*    attnp  = (float*)take((size_t)KCH * NH * LSEQ * LSEQ * sizeof(float));
  _Float16* a16    = (_Float16*)take((size_t)NH * LSEQ * LSEQ * sizeof(_Float16));
  _Float16* wt     = (_Float16*)take((size_t)5 * DP * DP * sizeof(_Float16));
  _Float16* og16   = pn16;   // pn dead after proj_kernel

  prep_weights_kernel<<<64, 256, 0, stream>>>(Wq, Wk, Wv, Wg, Wout, wt);
  ln_pair_kernel<<<NROW, 128, 0, stream>>>(pair, gamma_p, beta_p, pn16);
  ln_bias_kernel<<<NROW, 128, 0, stream>>>(bias, gamma_b, beta_b, Wb, bterm);
  proj_kernel<<<NROW / 16, 128, 0, stream>>>(pn16, wt, bg, qT, kT, vT, gate16);
  attn_kernel<<<dim3(6, 6, NH * KCH), 128, 0, stream>>>(qT, kT, attnp);
  softmax_kernel<<<NH * LSEQ, 128, 0, stream>>>(attnp, bterm, a16);
  av_kernel<<<dim3(LSEQ, NH), 128, 0, stream>>>(a16, vT, gate16, og16);
  out_kernel<<<NROW / 16 / 4, 128, 0, stream>>>(og16, wt + 4 * 16384, bout, out);
}
